// MoEJEPAPredictor_20813411516576
// MI455X (gfx1250) — matrix operand order repaired and checked by reading, not compiled, not run on hardware
//
#include <hip/hip_runtime.h>
#include <cstdint>
#include <cstddef>

// ---------------------------------------------------------------------------
// Types for WMMA fragments (gfx1250, wave32). v16bf = 16 bf16 = 8 VGPRs.
// ---------------------------------------------------------------------------
typedef __attribute__((ext_vector_type(16))) __bf16        v16bf;
typedef __attribute__((ext_vector_type(8)))  float         v8f;
typedef __attribute__((ext_vector_type(4)))  float         f4v;

union FragBF { v16bf v; f4v f[2]; };          // 32 bytes
union H8     { f4v f; unsigned short u[8]; }; // 16 bytes

__device__ __forceinline__ unsigned short f2bf(float x) {
  unsigned int u = __float_as_uint(x);
  u += 0x7FFFu + ((u >> 16) & 1u);            // round-to-nearest-even
  return (unsigned short)(u >> 16);
}
__device__ __forceinline__ float bf2f(unsigned short h) {
  return __uint_as_float(((unsigned int)h) << 16);
}
__device__ __forceinline__ float gelu_exact(float x) {
  return 0.5f * x * (1.f + erff(x * 0.70710678118654752440f));
}
__device__ __forceinline__ v8f wmma_bf16(v16bf a, v16bf b, v8f c) {
  return __builtin_amdgcn_wmma_f32_16x16x32_bf16(false, a, false, b, (short)0, c,
                                                 false, false);
}

// ---------------------------------------------------------------------------
// Generic bf16 WMMA GEMM: C[M,N] = act(A_bf16[M,K] @ B_f32[K,N] + bias (+resid))
// Block = 256 threads (8 waves). Block tile 64(M) x 128(N), K-step 32.
// Waves arranged 2(M) x 4(N); each wave owns a 32x32 output = 4 accumulators,
// issuing 4 independent WMMAs per K-step (pipelines past WMMA RAW hazards).
// Optional row gather for A (idxA) and row scatter for C (idxC).
// M may be overridden by a device counter (per-expert token counts).
// ---------------------------------------------------------------------------
__global__ __launch_bounds__(256)
void gemm_bf16_wmma(const unsigned short* __restrict__ A, int lda,
                    const float* __restrict__ Bw, int ldb,
                    const float* __restrict__ bias,
                    const float* __restrict__ resid, int ldres,
                    float* __restrict__ Cf, unsigned short* __restrict__ Cbf, int ldc,
                    int M, const int* __restrict__ Mdev, int Kdim,
                    const int* __restrict__ idxA, const int* __restrict__ idxC,
                    int idxBase, int doGelu)
{
  if (Mdev) M = *Mdev;                       // uniform scalar load
  const int mbase = blockIdx.y * 64;
  if (mbase >= M) return;                    // early-exit for empty expert tiles
  const int nbase = blockIdx.x * 128;

  __shared__ unsigned short sA[64 * 32];     // [m][k] bf16  (4 KB)
  __shared__ unsigned short sB[128 * 32];    // [n][k] bf16  (8 KB, transposed fill)

  const int tid  = threadIdx.x;
  const int lane = tid & 31;
  const int wave = tid >> 5;                 // 0..7
  const int wm   = wave & 1;                 // 2 waves along M (32 rows each)
  const int wn   = wave >> 1;                // 4 waves along N (32 cols each)
  const int l16  = lane & 15;
  const int hi   = lane >> 4;

  v8f acc[2][2];
  const v8f z8 = {};
  acc[0][0] = z8; acc[0][1] = z8; acc[1][0] = z8; acc[1][1] = z8;

  // A loader: 64 rows x 32 halves; each thread one 16-byte chunk.
  const int lm = tid >> 2;                   // 0..63
  const int lk = (tid & 3) * 8;              // 0,8,16,24
  long arow = -1;
  if (mbase + lm < M) {
    const int r = mbase + lm;
    arow = idxA ? (long)idxA[idxBase + r] : (long)r;
  }
  // B loader: 32 k-rows x 128 cols f32; each thread 16 cols of one k-row.
  const int bk  = tid >> 3;                  // 0..31
  const int bn0 = (tid & 7) * 16;

  for (int kb = 0; kb < Kdim; kb += 32) {
    {
      f4v av = {0.f, 0.f, 0.f, 0.f};
      if (arow >= 0) av = *(const f4v*)(A + arow * (long)lda + kb + lk);
      *(f4v*)(&sA[lm * 32 + lk]) = av;
    }
    const float* bp = Bw + (long)(kb + bk) * ldb + nbase + bn0;
#pragma unroll
    for (int j = 0; j < 4; ++j) {
      f4v f = *(const f4v*)(bp + j * 4);
      sB[(bn0 + j * 4 + 0) * 32 + bk] = f2bf(f.x);
      sB[(bn0 + j * 4 + 1) * 32 + bk] = f2bf(f.y);
      sB[(bn0 + j * 4 + 2) * 32 + bk] = f2bf(f.z);
      sB[(bn0 + j * 4 + 3) * 32 + bk] = f2bf(f.w);
    }
    // Prefetch next K-step's B rows into cache while we compute (gfx1250
    // global_prefetch_b8).
    if (kb + 32 < Kdim)
      __builtin_prefetch((const void*)(bp + (long)32 * ldb), 0, 1);
    __syncthreads();

    // A fragments: 16x32, lane l16 = row; lo lanes K 0-7/16-23, hi lanes +8
    FragBF fa[2], fb[2];
#pragma unroll
    for (int su = 0; su < 2; ++su) {
      const unsigned short* ar = &sA[(wm * 32 + su * 16 + l16) * 32];
      fa[su].f[0] = *(const f4v*)(ar + hi * 8);
      fa[su].f[1] = *(const f4v*)(ar + 16 + hi * 8);
    }
    // B fragments: 32x16, lane = column; lo lanes K 0-7/16-23, hi lanes +8
#pragma unroll
    for (int sn = 0; sn < 2; ++sn) {
      const unsigned short* br = &sB[(wn * 32 + sn * 16 + l16) * 32 + hi * 8];
      fb[sn].f[0] = *(const f4v*)(br);
      fb[sn].f[1] = *(const f4v*)(br + 16);
    }
#pragma unroll
    for (int su = 0; su < 2; ++su)
#pragma unroll
      for (int sn = 0; sn < 2; ++sn)
        acc[su][sn] = wmma_bf16(fa[su].v, fb[sn].v, acc[su][sn]);
    __syncthreads();
  }

  // Epilogue. C layout: VGPR r -> row (r + hi*8), col = lane&15.
#pragma unroll
  for (int sn = 0; sn < 2; ++sn) {
    const int col = nbase + wn * 32 + sn * 16 + l16;
    const float bv = bias ? bias[col] : 0.f;
#pragma unroll
    for (int su = 0; su < 2; ++su) {
#pragma unroll
      for (int r = 0; r < 8; ++r) {
        const int m = mbase + wm * 32 + su * 16 + r + hi * 8;
        if (m >= M) continue;
        const long crow = idxC ? (long)idxC[idxBase + m] : (long)m;
        float v = acc[su][sn][r] + bv;
        if (resid) v += resid[crow * (long)ldres + col];
        if (doGelu) v = gelu_exact(v);
        if (Cf)  Cf[crow * (long)ldc + col] = v;
        if (Cbf) Cbf[crow * (long)ldc + col] = f2bf(v);
      }
    }
  }
}

// ---------------------------------------------------------------------------
// Flash attention, one wave per (b, head, 16-row q tile). HD=64, S=513.
// qkv: T x 3072 bf16 (q|k|v, head-major 64-dim slices). obf: T x 1024 bf16.
// ---------------------------------------------------------------------------
__global__ __launch_bounds__(32)
void flash_attn_wmma(const unsigned short* __restrict__ qkv,
                     unsigned short* __restrict__ obf)
{
  const int qt = blockIdx.x, h = blockIdx.y, b = blockIdx.z;
  const int lane = threadIdx.x;
  const int l16 = lane & 15;
  const int hi  = lane >> 4;

  __shared__ unsigned short sP[16 * 32];     // [qrow][key] bf16
  __shared__ unsigned short sVt[64 * 32];    // [dim][key]  bf16

  const long rs = 3072;
  const unsigned short* qb  = qkv + (long)b * 513 * rs + h * 64;
  const unsigned short* kbp = qb + 1024;
  const unsigned short* vbp = qb + 2048;

  // Q fragments (held for the whole K loop): dims 0-31 and 32-63
  FragBF aq0, aq1;
  {
    const f4v z = {0.f, 0.f, 0.f, 0.f};
    aq0.f[0] = z; aq0.f[1] = z; aq1.f[0] = z; aq1.f[1] = z;
    const int qr = qt * 16 + l16;
    if (qr < 513) {
      const unsigned short* qp = qb + (long)qr * rs;
      aq0.f[0] = *(const f4v*)(qp + hi * 8);
      aq0.f[1] = *(const f4v*)(qp + 16 + hi * 8);
      aq1.f[0] = *(const f4v*)(qp + 32 + hi * 8);
      aq1.f[1] = *(const f4v*)(qp + 48 + hi * 8);
    }
  }

  float mrow[8], lrow[8];
  v8f oacc[4];
  const v8f z8 = {};
#pragma unroll
  for (int r = 0; r < 8; ++r) { mrow[r] = -1e30f; lrow[r] = 0.f; }
#pragma unroll
  for (int n = 0; n < 4; ++n) oacc[n] = z8;

  for (int kb = 0; kb < 513; kb += 32) {
    // ---- scores S = Q @ K^T for 32 keys (two 16-key groups) ----
    v8f sc[2];
#pragma unroll
    for (int g = 0; g < 2; ++g) {
      const int key = kb + g * 16 + l16;
      FragBF bk0, bk1;
      const f4v z = {0.f, 0.f, 0.f, 0.f};
      bk0.f[0] = z; bk0.f[1] = z; bk1.f[0] = z; bk1.f[1] = z;
      if (key < 513) {                       // key rows: dims contiguous -> direct B frag
        const unsigned short* kp = kbp + (long)key * rs;
        bk0.f[0] = *(const f4v*)(kp + hi * 8);
        bk0.f[1] = *(const f4v*)(kp + 16 + hi * 8);
        bk1.f[0] = *(const f4v*)(kp + 32 + hi * 8);
        bk1.f[1] = *(const f4v*)(kp + 48 + hi * 8);
      }
      v8f c = {};
      c = wmma_bf16(aq0.v, bk0.v, c);
      c = wmma_bf16(aq1.v, bk1.v, c);
      sc[g] = c;
    }
    // ---- scale + mask ----
#pragma unroll
    for (int g = 0; g < 2; ++g) {
      const bool valid = (kb + g * 16 + l16) < 513;
#pragma unroll
      for (int r = 0; r < 8; ++r)
        sc[g][r] = valid ? sc[g][r] * 0.125f : -1e30f;
    }
    // ---- online softmax (row = 16-lane half reductions) ----
    float alpha[8];
#pragma unroll
    for (int r = 0; r < 8; ++r) {
      float mx = fmaxf(sc[0][r], sc[1][r]);
#pragma unroll
      for (int off = 8; off > 0; off >>= 1)
        mx = fmaxf(mx, __shfl_xor(mx, off, 32));
      const float mnew = fmaxf(mrow[r], mx);
      alpha[r] = expf(mrow[r] - mnew);
      const float p0 = expf(sc[0][r] - mnew);
      const float p1 = expf(sc[1][r] - mnew);
      sc[0][r] = p0; sc[1][r] = p1;
      float rsum = p0 + p1;
#pragma unroll
      for (int off = 8; off > 0; off >>= 1)
        rsum += __shfl_xor(rsum, off, 32);
      lrow[r] = lrow[r] * alpha[r] + rsum;
      mrow[r] = mnew;
    }
#pragma unroll
    for (int n = 0; n < 4; ++n)
#pragma unroll
      for (int r = 0; r < 8; ++r) oacc[n][r] *= alpha[r];

    // ---- stage P (C-layout -> LDS row-major) and V^T ----
#pragma unroll
    for (int g = 0; g < 2; ++g)
#pragma unroll
      for (int r = 0; r < 8; ++r)
        sP[(r + hi * 8) * 32 + g * 16 + l16] = f2bf(sc[g][r]);
    {
      const int tok = kb + lane;
      if (tok < 513) {
        const unsigned short* vp = vbp + (long)tok * rs;
#pragma unroll
        for (int d8 = 0; d8 < 8; ++d8) {
          H8 hv; hv.f = *(const f4v*)(vp + d8 * 8);
#pragma unroll
          for (int j = 0; j < 8; ++j) sVt[(d8 * 8 + j) * 32 + lane] = hv.u[j];
        }
      } else {
#pragma unroll
        for (int d = 0; d < 64; ++d) sVt[d * 32 + lane] = 0;
      }
    }
    __syncthreads();

    // ---- O += P @ V  (A = P 16x32, B = V 32x16 per 16-dim slice) ----
    FragBF ap;
    ap.f[0] = *(const f4v*)(&sP[l16 * 32 + hi * 8]);
    ap.f[1] = *(const f4v*)(&sP[l16 * 32 + 16 + hi * 8]);
#pragma unroll
    for (int n = 0; n < 4; ++n) {
      FragBF bv;
      const unsigned short* vr = &sVt[(n * 16 + l16) * 32 + hi * 8];
      bv.f[0] = *(const f4v*)(vr);
      bv.f[1] = *(const f4v*)(vr + 16);
      oacc[n] = wmma_bf16(ap.v, bv.v, oacc[n]);
    }
    __syncthreads();
  }

#pragma unroll
  for (int r = 0; r < 8; ++r) {
    const int qr = qt * 16 + r + hi * 8;
    if (qr >= 513) continue;
    const float inv = 1.f / lrow[r];
    unsigned short* op = obf + ((long)(b * 513 + qr)) * 1024 + h * 64;
#pragma unroll
    for (int n = 0; n < 4; ++n)
      op[n * 16 + l16] = f2bf(oacc[n][r] * inv);
  }
}

// ---------------------------------------------------------------------------
// LayerNorm (D=1024) f32 -> bf16, one block per row.
// ---------------------------------------------------------------------------
__global__ __launch_bounds__(256)
void ln_to_bf16(const float* __restrict__ X, const float* __restrict__ gam,
                const float* __restrict__ bet, unsigned short* __restrict__ Y)
{
  __shared__ float red[256];
  __shared__ float stats[2];
  const long row = blockIdx.x;
  const float* x = X + row * 1024;
  float s = 0.f, ss = 0.f;
  for (int i = threadIdx.x; i < 1024; i += 256) { const float v = x[i]; s += v; ss += v * v; }
  red[threadIdx.x] = s; __syncthreads();
  for (int o = 128; o > 0; o >>= 1) { if (threadIdx.x < o) red[threadIdx.x] += red[threadIdx.x + o]; __syncthreads(); }
  if (threadIdx.x == 0) stats[0] = red[0];
  __syncthreads();
  red[threadIdx.x] = ss; __syncthreads();
  for (int o = 128; o > 0; o >>= 1) { if (threadIdx.x < o) red[threadIdx.x] += red[threadIdx.x + o]; __syncthreads(); }
  if (threadIdx.x == 0) stats[1] = red[0];
  __syncthreads();
  const float mean = stats[0] * (1.f / 1024.f);
  const float var  = stats[1] * (1.f / 1024.f) - mean * mean;
  const float inv  = rsqrtf(var + 1e-5f);
  unsigned short* y = Y + row * 1024;
  for (int i = threadIdx.x; i < 1024; i += 256)
    y[i] = f2bf((x[i] - mean) * inv * gam[i] + bet[i]);
}

// ---------------------------------------------------------------------------
// Action token: LN(action @ act_w + act_b) + dom_emb[domain_id] -> x row b*S.
// ---------------------------------------------------------------------------
__global__ __launch_bounds__(256)
void action_token_kernel(const float* __restrict__ action, const float* __restrict__ act_w,
                         const float* __restrict__ act_b, const float* __restrict__ g,
                         const float* __restrict__ be, const float* __restrict__ dom_emb,
                         const int* __restrict__ domain_id, float* __restrict__ x)
{
  const int b = blockIdx.x;
  __shared__ float vals[1024];
  __shared__ float red[256];
  __shared__ float stats[2];
  float av[9];
#pragma unroll
  for (int a = 0; a < 9; ++a) av[a] = action[b * 9 + a];
  for (int d = threadIdx.x; d < 1024; d += 256) {
    float acc = act_b[d];
#pragma unroll
    for (int a = 0; a < 9; ++a) acc += av[a] * act_w[a * 1024 + d];
    vals[d] = acc;
  }
  __syncthreads();
  float s = 0.f, ss = 0.f;
  for (int d = threadIdx.x; d < 1024; d += 256) { const float v = vals[d]; s += v; ss += v * v; }
  red[threadIdx.x] = s; __syncthreads();
  for (int o = 128; o > 0; o >>= 1) { if (threadIdx.x < o) red[threadIdx.x] += red[threadIdx.x + o]; __syncthreads(); }
  if (threadIdx.x == 0) stats[0] = red[0];
  __syncthreads();
  red[threadIdx.x] = ss; __syncthreads();
  for (int o = 128; o > 0; o >>= 1) { if (threadIdx.x < o) red[threadIdx.x] += red[threadIdx.x + o]; __syncthreads(); }
  if (threadIdx.x == 0) stats[1] = red[0];
  __syncthreads();
  const float mean = stats[0] * (1.f / 1024.f);
  const float var  = stats[1] * (1.f / 1024.f) - mean * mean;
  const float inv  = rsqrtf(var + 1e-5f);
  const int dom = domain_id[b];
  float* out = x + (long)b * 513 * 1024;
  for (int d = threadIdx.x; d < 1024; d += 256)
    out[d] = (vals[d] - mean) * inv * g[d] + be[d] + dom_emb[dom * 1024 + d];
}

// ---------------------------------------------------------------------------
// Router: probs=softmax(t @ W_r), top-2, combine weights, per-expert lists,
// deterministic per-block probs reduction for the aux loss.
// ---------------------------------------------------------------------------
__global__ __launch_bounds__(256)
void router_topk_kernel(const unsigned short* __restrict__ tbf, const float* __restrict__ rw,
                        float* __restrict__ topw, int* __restrict__ cnt,
                        int* __restrict__ toklist, int* __restrict__ scat,
                        float* __restrict__ psum_part)
{
  __shared__ float sp[256][8];
  const int t = blockIdx.x * 256 + threadIdx.x;
  float p[8];
#pragma unroll
  for (int e = 0; e < 8; ++e) p[e] = 0.f;
  if (t < 4104) {
    float acc[8];
#pragma unroll
    for (int e = 0; e < 8; ++e) acc[e] = 0.f;
    const unsigned short* tr = tbf + (long)t * 1024;
    for (int d0 = 0; d0 < 1024; d0 += 8) {
      H8 hv; hv.f = *(const f4v*)(tr + d0);
#pragma unroll
      for (int j = 0; j < 8; ++j) {
        const float hx = bf2f(hv.u[j]);
        const float* r = rw + (d0 + j) * 8;
#pragma unroll
        for (int e = 0; e < 8; ++e) acc[e] += hx * r[e];
      }
    }
    float mx = acc[0];
#pragma unroll
    for (int e = 1; e < 8; ++e) mx = fmaxf(mx, acc[e]);
    float se = 0.f;
#pragma unroll
    for (int e = 0; e < 8; ++e) { p[e] = expf(acc[e] - mx); se += p[e]; }
    const float isum = 1.f / se;
#pragma unroll
    for (int e = 0; e < 8; ++e) p[e] *= isum;
    int i0 = 0;
#pragma unroll
    for (int e = 1; e < 8; ++e) if (p[e] > p[i0]) i0 = e;
    int i1 = (i0 == 0) ? 1 : 0;
#pragma unroll
    for (int e = 0; e < 8; ++e) if (e != i0 && e != i1 && p[e] > p[i1]) i1 = e;
    const float wsum = p[i0] + p[i1];
    topw[t * 2 + 0] = p[i0] / wsum;
    topw[t * 2 + 1] = p[i1] / wsum;
    int pos = atomicAdd(&cnt[i0], 1);
    toklist[i0 * 4104 + pos] = t; scat[i0 * 4104 + pos] = t * 2 + 0;
    pos = atomicAdd(&cnt[i1], 1);
    toklist[i1 * 4104 + pos] = t; scat[i1 * 4104 + pos] = t * 2 + 1;
  }
#pragma unroll
  for (int e = 0; e < 8; ++e) sp[threadIdx.x][e] = p[e];
  __syncthreads();
  for (int o = 128; o > 0; o >>= 1) {
    if (threadIdx.x < o)
#pragma unroll
      for (int e = 0; e < 8; ++e) sp[threadIdx.x][e] += sp[threadIdx.x + o][e];
    __syncthreads();
  }
  if (threadIdx.x == 0)
#pragma unroll
    for (int e = 0; e < 8; ++e) psum_part[blockIdx.x * 8 + e] = sp[0][e];
}

__global__ void aux_zero_kernel(float* aux) { *aux = 0.f; }

__global__ void aux_finalize_kernel(const float* __restrict__ psum_part,
                                    const int* __restrict__ cnt, float* __restrict__ aux)
{
  float pm[8];
#pragma unroll
  for (int e = 0; e < 8; ++e) pm[e] = 0.f;
  for (int b = 0; b < 17; ++b)
#pragma unroll
    for (int e = 0; e < 8; ++e) pm[e] += psum_part[b * 8 + e];
  float a = 0.f;
#pragma unroll
  for (int e = 0; e < 8; ++e)
    a += ((float)cnt[e] / (float)(4104 * 2)) * (pm[e] / 4104.f);
  *aux += 8.f * a * 0.01f;
}

__global__ __launch_bounds__(256)
void moe_combine_kernel(const float* __restrict__ slotbuf, const float* __restrict__ topw,
                        float* __restrict__ x)
{
  const long i = (long)blockIdx.x * 256 + threadIdx.x;
  if (i >= (long)4104 * 1024) return;
  const long t = i >> 10;
  const int  d = (int)(i & 1023);
  x[i] += topw[t * 2 + 0] * slotbuf[(t * 2 + 0) * 1024 + d]
        + topw[t * 2 + 1] * slotbuf[(t * 2 + 1) * 1024 + d];
}

__global__ __launch_bounds__(256)
void scatter_particles_kernel(const float* __restrict__ p, float* __restrict__ x)
{
  const long i = (long)blockIdx.x * 256 + threadIdx.x;
  if (i >= (long)8 * 512 * 1024) return;
  const int  d  = (int)(i & 1023);
  const long kk = (i >> 10) & 511;
  const long b  = i >> 19;
  x[((b * 513) + 1 + kk) * 1024 + d] = p[i];
}

__global__ void build_pidx_kernel(int* __restrict__ pidx)
{
  const int i = blockIdx.x * 256 + threadIdx.x;
  if (i >= 4096) return;
  pidx[i] = (i >> 9) * 513 + 1 + (i & 511);
}

// ---------------------------------------------------------------------------
// Host-side orchestration (graph-capture safe: kernels + hipMemsetAsync only).
// ---------------------------------------------------------------------------
extern "C" void kernel_launch(void* const* d_in, const int* in_sizes, int n_in,
                              void* d_out, int out_size, void* d_ws, size_t ws_size,
                              hipStream_t stream)
{
  (void)in_sizes; (void)n_in; (void)out_size; (void)ws_size;
  const float* particles = (const float*)d_in[0];
  const float* action    = (const float*)d_in[1];
  const float* act_w     = (const float*)d_in[2];
  const float* act_b     = (const float*)d_in[3];
  const float* act_ln_g  = (const float*)d_in[4];
  const float* act_ln_b  = (const float*)d_in[5];
  const float* dom_emb   = (const float*)d_in[6];
  const float* router_w  = (const float*)d_in[7];
  const float* ln1_g     = (const float*)d_in[8];
  const float* ln1_b     = (const float*)d_in[9];
  const float* ln2_g     = (const float*)d_in[10];
  const float* ln2_b     = (const float*)d_in[11];
  const float* qkv_w     = (const float*)d_in[12];
  const float* qkv_b     = (const float*)d_in[13];
  const float* aow       = (const float*)d_in[14];
  const float* aob       = (const float*)d_in[15];
  const float* e1w       = (const float*)d_in[16];
  const float* e1b       = (const float*)d_in[17];
  const float* e2w       = (const float*)d_in[18];
  const float* e2b       = (const float*)d_in[19];
  const float* olng      = (const float*)d_in[20];
  const float* olnb      = (const float*)d_in[21];
  const float* proj_w    = (const float*)d_in[22];
  const float* proj_b    = (const float*)d_in[23];
  const int*   domain_id = (const int*)d_in[24];

  constexpr int  T = 4104, D = 1024, N3 = 3072, F = 2048, E = 8;
  constexpr int  MT = 65;                    // ceil(T / 64) M-tiles
  constexpr long ZS = (long)8 * 512 * 1024;

  char* w = (char*)d_ws;
  auto carve = [&](size_t bytes) -> char* {
    char* p = w; w += (bytes + 255) & ~(size_t)255; return p;
  };
  float*          x       = (float*)carve((size_t)T * D * 4);
  unsigned short* hbf     = (unsigned short*)carve((size_t)T * D * 2);
  unsigned short* qkvbf   = (unsigned short*)carve((size_t)T * N3 * 2);
  unsigned short* obf     = (unsigned short*)carve((size_t)T * D * 2);
  unsigned short* hh      = (unsigned short*)carve((size_t)T * 2 * F * 2);
  float*          slotbuf = (float*)carve((size_t)T * 2 * D * 4);
  float*          topw    = (float*)carve((size_t)T * 2 * 4);
  int*            cnt     = (int*)carve((size_t)E * 4);
  int*            toklist = (int*)carve((size_t)E * T * 4);
  int*            scat    = (int*)carve((size_t)E * T * 4);
  float*          psum    = (float*)carve((size_t)17 * E * 4);
  int*            pidx    = (int*)carve((size_t)4096 * 4);

  float* outz = (float*)d_out;
  float* auxp = outz + ZS;

  aux_zero_kernel<<<1, 1, 0, stream>>>(auxp);
  scatter_particles_kernel<<<(int)(((long)8 * 512 * 1024 + 255) / 256), 256, 0, stream>>>(particles, x);
  action_token_kernel<<<8, 256, 0, stream>>>(action, act_w, act_b, act_ln_g, act_ln_b,
                                             dom_emb, domain_id, x);

  for (int l = 0; l < 2; ++l) {
    ln_to_bf16<<<T, 256, 0, stream>>>(x, ln1_g + l * D, ln1_b + l * D, hbf);
    gemm_bf16_wmma<<<dim3(N3 / 128, MT), 256, 0, stream>>>(
        hbf, D, qkv_w + (long)l * D * N3, N3, qkv_b + l * N3,
        nullptr, 0, nullptr, qkvbf, N3, T, nullptr, D, nullptr, nullptr, 0, 0);
    flash_attn_wmma<<<dim3(33, 16, 8), 32, 0, stream>>>(qkvbf, obf);
    gemm_bf16_wmma<<<dim3(D / 128, MT), 256, 0, stream>>>(
        obf, D, aow + (long)l * D * D, D, aob + l * D,
        x, D, x, nullptr, D, T, nullptr, D, nullptr, nullptr, 0, 0);
    ln_to_bf16<<<T, 256, 0, stream>>>(x, ln2_g + l * D, ln2_b + l * D, hbf);
    hipMemsetAsync(cnt, 0, E * 4, stream);
    router_topk_kernel<<<17, 256, 0, stream>>>(hbf, router_w, topw, cnt, toklist, scat, psum);
    aux_finalize_kernel<<<1, 1, 0, stream>>>(psum, cnt, auxp);
    for (int e = 0; e < E; ++e) {
      gemm_bf16_wmma<<<dim3(F / 128, MT), 256, 0, stream>>>(
          hbf, D, e1w + (long)(l * E + e) * D * F, F, e1b + (long)(l * E + e) * F,
          nullptr, 0, nullptr, hh, F, T, cnt + e, D, toklist, scat, e * T, 1);
      gemm_bf16_wmma<<<dim3(D / 128, MT), 256, 0, stream>>>(
          hh, F, e2w + (long)(l * E + e) * F * D, D, e2b + (long)(l * E + e) * D,
          nullptr, 0, slotbuf, nullptr, D, T, cnt + e, F, scat, scat, e * T, 0);
    }
    moe_combine_kernel<<<(int)(((long)T * D + 255) / 256), 256, 0, stream>>>(slotbuf, topw, x);
  }

  ln_to_bf16<<<T, 256, 0, stream>>>(x, olng, olnb, hbf);
  build_pidx_kernel<<<16, 256, 0, stream>>>(pidx);
  gemm_bf16_wmma<<<dim3(D / 128, 64), 256, 0, stream>>>(
      hbf, D, proj_w, D, proj_b, nullptr, 0, outz, nullptr, D, 4096, nullptr, D,
      pidx, nullptr, 0, 0);
}
